// KerasLIFLayerSparse_75952201662775
// MI455X (gfx1250) — matrix operand order repaired and checked by reading, not compiled, not run on hardware
//
#include <hip/hip_runtime.h>
#include <hip/hip_bf16.h>

typedef __attribute__((ext_vector_type(16))) _Float16 v16h;
typedef __attribute__((ext_vector_type(8)))  _Float16 v8h;
typedef __attribute__((ext_vector_type(8)))  float    v8f;

constexpr int SEQ_LEN = 1000;
constexpr int BATCH   = 32;
constexpr int UNITS   = 1024;
constexpr int IN_DIM  = 1024;
constexpr int K_IN    = 32;
constexpr int SOUT    = 32;
constexpr int ROWS    = SEQ_LEN * BATCH;   // 32000 time-batch rows
constexpr int XPAD    = 1032;              // 16B-aligned row stride, 4-dword bank stagger
constexpr int MROWS   = 32;                // rows (time-batch) per block: 2 M-tiles

// ---------------------------------------------------------------------------
// Kernel 1: convert weights f32 -> f16 (row-major [UNITS, IN_DIM]), 2MB, L2-resident.
// ---------------------------------------------------------------------------
__global__ void cvt_w_kernel(const float* __restrict__ w, _Float16* __restrict__ wh) {
    int i = blockIdx.x * blockDim.x + threadIdx.x;
    wh[i] = (_Float16)w[i];
}

// ---------------------------------------------------------------------------
// Kernel 2: SYN[row, u] = sum_i X[row, i] * w[u, i] via v_wmma_f32_16x16x32_f16.
// Block = 256 threads (8 waves). Block tile: 32 rows x 128 units.
// Each wave computes two 16x16 M-tiles, reusing every B fragment twice
// (halves the L2 weight-fragment traffic, which dominates the GEMM cost).
// X tile (32 x IN_DIM multi-hot spike counts, f16-exact) staged in LDS.
// Output written straight into the `states` region of d_out.
// ---------------------------------------------------------------------------
__global__ __launch_bounds__(256)
void syn_gemm_kernel(const _Float16* __restrict__ wh,
                     const float*    __restrict__ spike_ids,   // [ROWS, K_IN] f32 ids
                     const int*      __restrict__ num_spikes,  // [ROWS] i32
                     float*          __restrict__ syn_out) {   // [ROWS, UNITS]
    __shared__ _Float16 Xs[MROWS][XPAD];

    const int tid     = threadIdx.x;
    const int lane    = tid & 31;
    const int wid     = tid >> 5;
    const int rowBase = blockIdx.y * MROWS;
    const int u0      = blockIdx.x * 128 + wid * 16;

    // Zero the X tile (dword stores; XPAD is even so rows stay dword-aligned).
    for (int i = tid; i < MROWS * XPAD / 2; i += 256)
        ((unsigned int*)Xs)[i] = 0u;
    __syncthreads();

    // Scatter spike ids: one thread per row (duplicates accumulate, matching einsum).
    if (tid < MROWS) {
        const int row = rowBase + tid;
        int num = num_spikes[row];
        num = num < 0 ? 0 : (num > K_IN ? K_IN : num);
        const float* ids = spike_ids + (size_t)row * K_IN;
        for (int k = 0; k < num; ++k) {
            int id = ((int)ids[k]) & (IN_DIM - 1);
            Xs[tid][id] = Xs[tid][id] + (_Float16)1.0f;
        }
    }
    __syncthreads();

    const int m = lane & 15;   // A row / B column / D column within the 16x16 tile
    const int g = lane >> 4;   // lane group

    v8f c0 = {};
    v8f c1 = {};
    const _Float16* bptr = wh + (size_t)(u0 + m) * IN_DIM + 16 * g;

    for (int ib = 0; ib < IN_DIM; ib += 32) {
        // B fragment: lane holds column N=m, K rows 16g..16g+15 (contiguous 32B)
        v16h b = *(const v16h*)(bptr + ib);

        // A fragments: lane holds row M=m, K rows 16g..16g+15 (contiguous)
        v8h lo0 = *(const v8h*)&Xs[m][ib + 16 * g];
        v8h hi0 = *(const v8h*)&Xs[m][ib + 16 * g + 8];
        v16h a0 = __builtin_shufflevector(lo0, hi0,
                                          0, 1, 2, 3, 4, 5, 6, 7,
                                          8, 9, 10, 11, 12, 13, 14, 15);
        v8h lo1 = *(const v8h*)&Xs[16 + m][ib + 16 * g];
        v8h hi1 = *(const v8h*)&Xs[16 + m][ib + 16 * g + 8];
        v16h a1 = __builtin_shufflevector(lo1, hi1,
                                          0, 1, 2, 3, 4, 5, 6, 7,
                                          8, 9, 10, 11, 12, 13, 14, 15);

        c0 = __builtin_amdgcn_wmma_f32_16x16x32_f16(false, a0, false, b,
                                                    (short)0, c0, false, false);
        c1 = __builtin_amdgcn_wmma_f32_16x16x32_f16(false, a1, false, b,
                                                    (short)0, c1, false, false);
    }

    // D layout: VGPR r -> row r + 8g, lane -> column m.
#pragma unroll
    for (int r = 0; r < 8; ++r) {
        syn_out[(size_t)(rowBase + r + 8 * g) * UNITS + u0 + m]      = c0[r];
        syn_out[(size_t)(rowBase + 16 + r + 8 * g) * UNITS + u0 + m] = c1[r];
    }
}

// ---------------------------------------------------------------------------
// Kernel 3: sequential LIF scan. One block per batch, 256 threads x 4 units.
// Reads SYN from the states region and overwrites it in place with v.
// Ascending first-32 spiking ids via block-wide prefix sum (thread order = id order).
// ---------------------------------------------------------------------------
__global__ __launch_bounds__(256)
void lif_scan_kernel(const float* __restrict__ init_state,
                     const float* __restrict__ decay,
                     const float* __restrict__ thresh,
                     float* __restrict__ out_ids,   // [SEQ, BATCH, SOUT]
                     float* __restrict__ out_num,   // [SEQ, BATCH]
                     float* states) {               // in: SYN, out: v
    const int b    = blockIdx.x;
    const int tid  = threadIdx.x;
    const int lane = tid & 31;
    const int wid  = tid >> 5;
    const int u0   = tid * 4;

    __shared__ int wsum[8];

    const float4 dec = *(const float4*)&decay[u0];
    const float4 th  = *(const float4*)&thresh[u0];
    const float4 omd = make_float4(1.0f - dec.x, 1.0f - dec.y,
                                   1.0f - dec.z, 1.0f - dec.w);
    float4 st = *(const float4*)&init_state[(size_t)b * UNITS + u0];

    for (int t = 0; t < SEQ_LEN; ++t) {
        const size_t idx = ((size_t)t * BATCH + b) * UNITS + u0;
        const float4 syn = *(const float4*)&states[idx];

        float4 v;
        v.x = dec.x * st.x + omd.x * syn.x;
        v.y = dec.y * st.y + omd.y * syn.y;
        v.z = dec.z * st.z + omd.z * syn.z;
        v.w = dec.w * st.w + omd.w * syn.w;
        *(float4*)&states[idx] = v;   // pre-reset membrane potential output

        const int s0 = v.x > th.x, s1 = v.y > th.y, s2 = v.z > th.z, s3 = v.w > th.w;
        st.x = s0 ? 0.0f : v.x;
        st.y = s1 ? 0.0f : v.y;
        st.z = s2 ? 0.0f : v.z;
        st.w = s3 ? 0.0f : v.w;

        const int cnt = s0 + s1 + s2 + s3;

        // Wave-level inclusive prefix sum of per-thread spike counts.
        int inc = cnt;
#pragma unroll
        for (int off = 1; off < 32; off <<= 1) {
            int o = __shfl_up(inc, off, 32);
            if (lane >= off) inc += o;
        }
        if (lane == 31) wsum[wid] = inc;
        __syncthreads();

        int offset = 0, total = 0;
#pragma unroll
        for (int w2 = 0; w2 < 8; ++w2) {
            const int s = wsum[w2];
            offset += (w2 < wid) ? s : 0;
            total  += s;
        }
        int pos = offset + inc - cnt;                  // exclusive prefix, thread order
        const int cmax = total < SOUT ? total : SOUT;

        float* oid = out_ids + ((size_t)t * BATCH + b) * SOUT;
        if (s0) { if (pos < SOUT) oid[pos] = (float)(u0 + 0); ++pos; }
        if (s1) { if (pos < SOUT) oid[pos] = (float)(u0 + 1); ++pos; }
        if (s2) { if (pos < SOUT) oid[pos] = (float)(u0 + 2); ++pos; }
        if (s3) { if (pos < SOUT) oid[pos] = (float)(u0 + 3); ++pos; }
        if (tid < SOUT && tid >= cmax) oid[tid] = 0.0f; // zero padding (disjoint slots)
        if (tid == 0) out_num[(size_t)t * BATCH + b] = (float)cmax;

        __syncthreads();
    }
}

// ---------------------------------------------------------------------------
extern "C" void kernel_launch(void* const* d_in, const int* in_sizes, int n_in,
                              void* d_out, int out_size, void* d_ws, size_t ws_size,
                              hipStream_t stream) {
    const float* w          = (const float*)d_in[0];
    const float* init_state = (const float*)d_in[1];
    const float* spike_ids  = (const float*)d_in[2];
    const int*   num_spikes = (const int*)d_in[3];
    const float* decay      = (const float*)d_in[4];
    const float* thresh     = (const float*)d_in[5];

    float* out      = (float*)d_out;
    float* out_ids  = out;                                         // [1000,32,32]
    float* out_num  = out + (size_t)SEQ_LEN * BATCH * SOUT;        // [1000,32,1]
    float* states   = out_num + (size_t)SEQ_LEN * BATCH;           // [1000,32,1024]

    _Float16* wh = (_Float16*)d_ws;                                // 2 MB f16 weights

    cvt_w_kernel<<<(UNITS * IN_DIM) / 256, 256, 0, stream>>>(w, wh);

    dim3 ggrid(UNITS / 128, ROWS / MROWS);                         // 8 x 1000 blocks
    syn_gemm_kernel<<<ggrid, 256, 0, stream>>>(wh, spike_ids, num_spikes, states);

    lif_scan_kernel<<<BATCH, 256, 0, stream>>>(init_state, decay, thresh,
                                               out_ids, out_num, states);
}
